// SparseLSTM_76845554860471
// MI455X (gfx1250) — matrix operand order repaired and checked by reading, not compiled, not run on hardware
//
#include <hip/hip_runtime.h>
#include <stdint.h>

// ---------------------------------------------------------------------------
// SparseLSTM on gfx1250 (MI455X), wave32 + WMMA bf16.
//
// Fused design: no xg materialization. 32 persistent workgroups = 4 blocks x
// 8 slices; each WG owns 32 hidden units (all 4 gates, gate-interleaved rows)
// with its 128x768 bf16 weight slice resident in LDS for the whole sequence.
// Per step: [64 x 768] @ [768 x 128]^T via v_wmma_f32_16x16x32_bf16, local
// elementwise LSTM update, one device-scope barrier per block per step with
// ping-pong h buffers in L2.
// ---------------------------------------------------------------------------

typedef __attribute__((ext_vector_type(16))) __bf16        v16bf;
typedef __attribute__((ext_vector_type(8)))  float         v8f;
typedef __attribute__((ext_vector_type(4)))  float         f4;   // POD, union-safe
typedef __attribute__((ext_vector_type(4)))  unsigned short us4;

union ABfrag { f4 q[2]; v16bf v; };

__device__ __forceinline__ unsigned short f2bf(float x) {
  unsigned u = __float_as_uint(x);
  u += 0x7FFFu + ((u >> 16) & 1u);      // round-to-nearest-even
  return (unsigned short)(u >> 16);
}
__device__ __forceinline__ float sigm(float x) { return 1.0f / (1.0f + __expf(-x)); }

// ---------------- workspace layout (bytes) ----------------
#define SZ_XK    (4ull * 512 * 64 * 512 * 2)   // per-block bf16 x slices: 134,217,728
#define OFF_XK   0ull
#define OFF_WCAT (OFF_XK + SZ_XK)              // transposed wg weight slices [32][768][128] bf16
#define SZ_WCAT  (32ull * 768 * 128 * 2)       // 6,291,456
#define OFF_BIAS (OFF_WCAT + SZ_WCAT)          // [32][128] f32
#define SZ_BIAS  (4096ull * 4)
#define OFF_H0   (OFF_BIAS + SZ_BIAS)          // h ping  [4][64][256] bf16
#define SZ_H     (65536ull * 2)
#define OFF_H1   (OFF_H0 + SZ_H)               // h pong
#define OFF_CNT  (OFF_H1 + SZ_H)               // 4 arrival counters (+pad)

// LDS layout (bytes) inside the persistent kernel
#define LDS_W    0                              // [768][136] halfs, 272B pitch (conflict-free)
#define LDS_GSC  208896                         // [128][65] f32 gate scratch
#define LDS_CST  242176                         // [32][66]  f32 cell state
#define LDS_TOT  250624

#define OUT_H    33554432ull                    // T*B*1024
#define OUT_C    (33554432ull + 65536ull)

// --------- prep kernels (run once per launch, all cheap/streaming) ---------

// x[T,B,1024] f32 -> per-block contiguous bf16 slices xk[k][t][b][512]
__global__ void k_convert_x(const float* __restrict__ x, unsigned short* __restrict__ xk) {
  const int xstart[4] = {0, 170, 341, 512};
  size_t l = ((size_t)blockIdx.x * 256 + threadIdx.x) * 4;  // element index, total 2^26
  int j = (int)(l & 511);
  int b = (int)((l >> 9) & 63);
  int t = (int)((l >> 15) & 511);
  int k = (int)(l >> 24);
  const float* src = x + ((size_t)t * 64 + b) * 1024 + xstart[k] + j;
  us4 o;
  o.x = f2bf(src[0]); o.y = f2bf(src[1]); o.z = f2bf(src[2]); o.w = f2bf(src[3]);
  *(us4*)(xk + l) = o;
}

// W_ih[k][g][512], W_hh[k][g][256] -> wcat[wg][col(0..767)][row(0..127)] bf16
// row r = unit*4 + gate (gate-interleaved so each WG owns i,f,g,o of its units)
// col < 512 -> input weight, col >= 512 -> hidden weight. Stored K-major
// (transposed) to match the WMMA B-operand layout (lane = N, slots = K).
__global__ void k_prep_w(const float* __restrict__ Wih, const float* __restrict__ Whh,
                         unsigned short* __restrict__ wcat) {
  size_t idx = (size_t)blockIdx.x * 256 + threadIdx.x;   // < 32*768*128
  int row = (int)(idx & 127);
  int col = (int)((idx >> 7) % 768);
  int wg  = (int)(idx / (768 * 128));
  int k = wg >> 3, q = wg & 7;
  int g = (row & 3) * 256 + q * 32 + (row >> 2);
  float w = (col < 512) ? Wih[((size_t)k * 1024 + g) * 512 + col]
                        : Whh[((size_t)k * 1024 + g) * 256 + (col - 512)];
  wcat[idx] = f2bf(w);
}

// combined bias, initial h (bf16, block-major), barrier counters
__global__ void k_prep_small(const float* __restrict__ bih, const float* __restrict__ bhh,
                             const float* __restrict__ h0, float* __restrict__ bias,
                             unsigned short* __restrict__ hb0, unsigned* __restrict__ cnt) {
  int idx = blockIdx.x * 256 + threadIdx.x;
  if (idx < 4096) {
    int wg = idx >> 7, r = idx & 127;
    int k = wg >> 3, q = wg & 7;
    int g = (r & 3) * 256 + q * 32 + (r >> 2);
    bias[idx] = bih[k * 1024 + g] + bhh[k * 1024 + g];
  } else if (idx < 4096 + 65536) {
    int j = idx - 4096;                       // (k*64+b)*256 + jj
    int k = j >> 14, b = (j >> 8) & 63, jj = j & 255;
    hb0[j] = f2bf(h0[b * 1024 + k * 256 + jj]);
  } else if (idx < 4096 + 65536 + 8) {
    cnt[idx - 4096 - 65536] = 0;
  }
}

// ------------------------- persistent recurrence --------------------------

__global__ __launch_bounds__(256, 1)
void k_lstm(const unsigned short* __restrict__ xk,
            const unsigned short* __restrict__ wcat,
            const float* __restrict__ bias,
            unsigned short* __restrict__ hb0,
            unsigned short* __restrict__ hb1,
            unsigned* __restrict__ cnt,
            const float* __restrict__ c0,
            float* __restrict__ out) {
  extern __shared__ char smem[];
  float* gsc = (float*)(smem + LDS_GSC);
  float* cst = (float*)(smem + LDS_CST);

  const int tid = threadIdx.x;
  const int wg  = blockIdx.x;
  const int k = wg >> 3, q = wg & 7;
  const int lane = tid & 31;
  const int wv   = tid >> 5;
  const int ll = lane & 15;          // A: M row within tile / B: N column within tile
  const int lh = lane >> 4;          // half-group select
  const int m0 = (wv & 1) * 32;      // batch tile base (2 waves over M=64)
  const int n0 = (wv >> 1) * 32;     // gate-row tile base (4 waves over N=128)

  // Stage this WG's transposed weight slice [768][128] into LDS, 272B pitch.
  {
    const unsigned short* src = wcat + (size_t)wg * 768 * 128;
    for (int i = tid; i < 768 * 16; i += 256) {
      int r = i >> 4, c = i & 15;
      *(f4*)(smem + r * 272 + c * 16) =
          *(const f4*)((const char*)src + (size_t)r * 256 + c * 16);
    }
  }

  // Elementwise ownership: unit eu (0..31), batch run eb0..eb0+7.
  const int eu  = tid >> 3;
  const int eb0 = (tid & 7) * 8;
  const float bi  = bias[wg * 128 + eu * 4 + 0];
  const float bff = bias[wg * 128 + eu * 4 + 1];
  const float bgg = bias[wg * 128 + eu * 4 + 2];
  const float boo = bias[wg * 128 + eu * 4 + 3];
  for (int e = 0; e < 8; ++e) {
    int b = eb0 + e;
    cst[eu * 66 + b] = c0[b * 1024 + k * 256 + q * 32 + eu];
  }
  __syncthreads();

  const size_t xkbase = (size_t)k * 512 * 64 * 512;

  for (int t = 0; t < 512; ++t) {
    const unsigned short* hsrc = (t & 1) ? hb1 : hb0;
    unsigned short*       hdst = (t & 1) ? hb0 : hb1;

    v8f acc00 = {0,0,0,0,0,0,0,0};
    v8f acc01 = acc00, acc10 = acc00, acc11 = acc00;

    // ---- input contribution: K = 0..511 (streamed bf16 x slice) ----
    const unsigned short* xrow0 = xk + xkbase + ((size_t)t * 64 + (m0 + ll)) * 512;
    const unsigned short* xrow1 = xk + xkbase + ((size_t)t * 64 + (m0 + 16 + ll)) * 512;
    for (int ks = 0; ks < 16; ++ks) {
      int kk = ks * 32;
      ABfrag a0, a1, b0, b1;
      // A layout: lane=M; slots: lanes0-15 K{kk..kk+7, kk+16..kk+23},
      //                         lanes16-31 K{kk+8..15, kk+24..31}
      const unsigned short* p0 = xrow0 + kk + lh * 8;
      a0.q[0] = *(const f4*)p0;  a0.q[1] = *(const f4*)(p0 + 16);
      const unsigned short* p1 = xrow1 + kk + lh * 8;
      a1.q[0] = *(const f4*)p1;  a1.q[1] = *(const f4*)(p1 + 16);
      // B layout: lane=N (ll), slots = K in same order as A
      const unsigned short* bp = (const unsigned short*)smem + (size_t)kk * 136 + n0 + ll;
      for (int s = 0; s < 16; ++s) {
        int kr = (s < 8) ? (8 * lh + s) : (8 + 8 * lh + s);
        ((unsigned short*)&b0)[s] = bp[kr * 136];
        ((unsigned short*)&b1)[s] = bp[kr * 136 + 16];
      }
      acc00 = __builtin_amdgcn_wmma_f32_16x16x32_bf16(false, a0.v, false, b0.v, (short)0, acc00, false, false);
      acc01 = __builtin_amdgcn_wmma_f32_16x16x32_bf16(false, a0.v, false, b1.v, (short)0, acc01, false, false);
      acc10 = __builtin_amdgcn_wmma_f32_16x16x32_bf16(false, a1.v, false, b0.v, (short)0, acc10, false, false);
      acc11 = __builtin_amdgcn_wmma_f32_16x16x32_bf16(false, a1.v, false, b1.v, (short)0, acc11, false, false);
    }
    // ---- recurrent contribution: K = 512..767 (h in L2, ping-pong) ----
    const unsigned short* hrow0 = hsrc + ((size_t)(k * 64) + (m0 + ll)) * 256;
    const unsigned short* hrow1 = hsrc + ((size_t)(k * 64) + (m0 + 16 + ll)) * 256;
    for (int ks = 0; ks < 8; ++ks) {
      int kk = ks * 32;
      ABfrag a0, a1, b0, b1;
      const unsigned short* p0 = hrow0 + kk + lh * 8;
      a0.q[0] = *(const f4*)p0;  a0.q[1] = *(const f4*)(p0 + 16);
      const unsigned short* p1 = hrow1 + kk + lh * 8;
      a1.q[0] = *(const f4*)p1;  a1.q[1] = *(const f4*)(p1 + 16);
      const unsigned short* bp = (const unsigned short*)smem + (size_t)(512 + kk) * 136 + n0 + ll;
      for (int s = 0; s < 16; ++s) {
        int kr = (s < 8) ? (8 * lh + s) : (8 + 8 * lh + s);
        ((unsigned short*)&b0)[s] = bp[kr * 136];
        ((unsigned short*)&b1)[s] = bp[kr * 136 + 16];
      }
      acc00 = __builtin_amdgcn_wmma_f32_16x16x32_bf16(false, a0.v, false, b0.v, (short)0, acc00, false, false);
      acc01 = __builtin_amdgcn_wmma_f32_16x16x32_bf16(false, a0.v, false, b1.v, (short)0, acc01, false, false);
      acc10 = __builtin_amdgcn_wmma_f32_16x16x32_bf16(false, a1.v, false, b0.v, (short)0, acc10, false, false);
      acc11 = __builtin_amdgcn_wmma_f32_16x16x32_bf16(false, a1.v, false, b1.v, (short)0, acc11, false, false);
    }

    // ---- spill gates to LDS (D layout: lane=N, vgpr=M; 65-dword pitch) ----
    {
      int n_ = n0 + ll;
      int mA = m0 + lh * 8;
      int mB = m0 + 16 + lh * 8;
#pragma unroll
      for (int v = 0; v < 8; ++v) {
        gsc[n_ * 65 + mA + v]        = acc00[v];
        gsc[(n_ + 16) * 65 + mA + v] = acc01[v];
        gsc[n_ * 65 + mB + v]        = acc10[v];
        gsc[(n_ + 16) * 65 + mB + v] = acc11[v];
      }
    }
    __syncthreads();

    // ---- elementwise LSTM update (gates i,f,g,o all local to this WG) ----
#pragma unroll
    for (int e = 0; e < 8; ++e) {
      int b = eb0 + e;
      float gi = gsc[(eu * 4 + 0) * 65 + b] + bi;
      float gf = gsc[(eu * 4 + 1) * 65 + b] + bff;
      float gg = gsc[(eu * 4 + 2) * 65 + b] + bgg;
      float go = gsc[(eu * 4 + 3) * 65 + b] + boo;
      float cp = cst[eu * 66 + b];
      float cn = sigm(gf) * cp + sigm(gi) * tanhf(gg);
      float hn = sigm(go) * tanhf(cn);
      cst[eu * 66 + b] = cn;
      hdst[(size_t)(k * 64 + b) * 256 + q * 32 + eu] = f2bf(hn);
      size_t oidx = ((size_t)t * 64 + b) * 1024 + k * 256 + q * 32 + eu;
      out[oidx] = hn;
      if (t == 511) {
        size_t fi = (size_t)b * 1024 + k * 256 + q * 32 + eu;
        out[OUT_H + fi] = hn;
        out[OUT_C + fi] = cn;
      }
    }

    // ---- per-block barrier: release h(t+1), acquire all 8 slices ----
    __threadfence();
    __syncthreads();
    if (tid == 0) {
      __hip_atomic_fetch_add(&cnt[k], 1u, __ATOMIC_ACQ_REL, __HIP_MEMORY_SCOPE_AGENT);
      unsigned tgt = 8u * (unsigned)(t + 1);
      while (__hip_atomic_load(&cnt[k], __ATOMIC_ACQUIRE, __HIP_MEMORY_SCOPE_AGENT) < tgt)
        __builtin_amdgcn_s_sleep(1);
    }
    __syncthreads();
  }
}

// ------------------------------ launch -------------------------------------

extern "C" void kernel_launch(void* const* d_in, const int* in_sizes, int n_in,
                              void* d_out, int out_size, void* d_ws, size_t ws_size,
                              hipStream_t stream) {
  (void)in_sizes; (void)n_in; (void)out_size; (void)ws_size;
  const float* x   = (const float*)d_in[0];
  const float* h0  = (const float*)d_in[1];
  const float* c0  = (const float*)d_in[2];
  const float* Wih = (const float*)d_in[3];
  const float* Whh = (const float*)d_in[4];
  const float* bih = (const float*)d_in[5];
  const float* bhh = (const float*)d_in[6];
  float* out = (float*)d_out;
  char*  ws  = (char*)d_ws;

  unsigned short* xk   = (unsigned short*)(ws + OFF_XK);
  unsigned short* wcat = (unsigned short*)(ws + OFF_WCAT);
  float*          bias = (float*)(ws + OFF_BIAS);
  unsigned short* hb0  = (unsigned short*)(ws + OFF_H0);
  unsigned short* hb1  = (unsigned short*)(ws + OFF_H1);
  unsigned*       cnt  = (unsigned*)(ws + OFF_CNT);

  k_convert_x<<<65536, 256, 0, stream>>>(x, xk);
  k_prep_w<<<12288, 256, 0, stream>>>(Wih, Whh, wcat);
  k_prep_small<<<273, 256, 0, stream>>>(bih, bhh, h0, bias, hb0, cnt);

  (void)hipFuncSetAttribute((const void*)k_lstm,
                            hipFuncAttributeMaxDynamicSharedMemorySize, LDS_TOT);
  k_lstm<<<32, 256, LDS_TOT, stream>>>(xk, wcat, bias, hb0, hb1, cnt, c0, out);
}
